// LSTM_84791244357714
// MI455X (gfx1250) — matrix operand order repaired and checked by reading, not compiled, not run on hardware
//
#include <hip/hip_runtime.h>
#include <hip/hip_bf16.h>

#define B_ 64
#define T_ 128
#define H_ 512
#define V_ 10000

typedef __attribute__((ext_vector_type(16))) __bf16 v16bf;
typedef __attribute__((ext_vector_type(8)))  float  v8f;

union BV {
    uint4  u[2];
    v16bf  v;
};

__device__ __forceinline__ unsigned short f2bf(float f) {
    unsigned int x = __float_as_uint(f);
    unsigned int r = x + 0x7FFFu + ((x >> 16) & 1u);  // round-to-nearest-even
    return (unsigned short)(r >> 16);
}

__device__ __forceinline__ float sigmoidf(float v) {
    return 1.f / (1.f + __expf(-v));
}

// ---------------------------------------------------------------------------
// Pack the 4 recurrent weight matrices (512x512 fp32, row-major [k][n]) into
// bf16 WMMA-B layout: [gate][ntile(32)][kchunk(16)][512], where element
// idx = lane*16 + j  holds  W[k = kc*32 + 8*(lane/16) + (j%8) + 16*(j/8)][n = nt*16 + lane%16].
// Each consuming lane then loads its 32 bytes contiguously (2 x b128).
// ---------------------------------------------------------------------------
__global__ void pack_wh_kernel(const float* __restrict__ Wi,
                               const float* __restrict__ Wf,
                               const float* __restrict__ Wo,
                               const float* __restrict__ Wc,
                               unsigned short* __restrict__ out) {
    int e = blockIdx.x * 256 + threadIdx.x;
    if (e >= 4 * 32 * 16 * 512) return;
    int idx  = e & 511;
    int kc   = (e >> 9) & 15;
    int nt   = (e >> 13) & 31;
    int g    = e >> 18;
    int lane = idx >> 4;
    int j    = idx & 15;
    int k = kc * 32 + ((lane >> 4) << 3) + (j & 7) + ((j >> 3) << 4);
    int n = nt * 16 + (lane & 15);
    const float* W = (g == 0) ? Wi : (g == 1) ? Wf : (g == 2) ? Wo : Wc;
    out[e] = f2bf(W[k * H_ + n]);
}

// Same packing for W_hq (512 x 10000, row-major): [ntile(625)][kchunk(16)][512]
__global__ void pack_whq_kernel(const float* __restrict__ Whq,
                                unsigned short* __restrict__ out) {
    int e = blockIdx.x * 256 + threadIdx.x;
    if (e >= 625 * 16 * 512) return;
    int idx  = e & 511;
    int kc   = (e >> 9) & 15;
    int nt   = e >> 13;               // 0..624
    int lane = idx >> 4;
    int j    = idx & 15;
    int k = kc * 32 + ((lane >> 4) << 3) + (j & 7) + ((j >> 3) << 4);
    int n = nt * 16 + (lane & 15);
    out[e] = f2bf(Whq[k * V_ + n]);
}

// ---------------------------------------------------------------------------
// Recurrent phase: one persistent workgroup (32 wave32 waves on one WGP).
// LDS: h (bf16, 64KB) + c (fp32, 128KB) = 192KB of the 320KB WGP LDS.
// Wave w owns m-tile (w%4) and n-tiles (w/4)*4..+3; gates are processed
// sequentially (f, i, g, o) so only 4 WMMA accumulators + the stashed
// input-gate activations are live at once (~100 VGPRs -> no spills).
// ---------------------------------------------------------------------------
__global__ __launch_bounds__(1024) void lstm_recurrent_kernel(
    const int*   __restrict__ x,        // (B, T)
    const float* __restrict__ state0,   // (B, H)
    const float* __restrict__ cell0,    // (B, H)
    const float* __restrict__ Wxi, const float* __restrict__ Wxf,
    const float* __restrict__ Wxo, const float* __restrict__ Wxc,
    const float* __restrict__ bi,  const float* __restrict__ bf_,
    const float* __restrict__ bo,  const float* __restrict__ bc,
    const unsigned short* __restrict__ WhP,   // packed [4][32][16][512] bf16
    unsigned short* __restrict__ hHist,       // [T][B][H] bf16
    float* __restrict__ outState,             // (B, H) fp32
    float* __restrict__ outCell)              // (B, H) fp32
{
    extern __shared__ char smemRaw[];
    unsigned short (*hS)[H_] = (unsigned short (*)[H_])smemRaw;            // 64 KB
    float          (*cS)[H_] = (float (*)[H_])(smemRaw + B_ * H_ * 2);     // 128 KB

    const int tid  = threadIdx.x;
    const int lane = tid & 31;
    const int wave = tid >> 5;
    const int half = lane >> 4;
    const int ln   = lane & 15;
    const int m    = wave & 3;    // m-tile (rows m*16 .. m*16+15)
    const int ng   = wave >> 2;   // n-tile group: tiles ng*4 .. ng*4+3

    // Seed LDS h (fp32 -> bf16) and c (fp32)
    for (int e = tid; e < B_ * H_; e += 1024) {
        ((unsigned short*)hS)[e] = f2bf(state0[e]);
        ((float*)cS)[e]          = cell0[e];
    }
    __syncthreads();

    const float* const WxArr[4] = {Wxi, Wxf, Wxo, Wxc};
    const float* const bArr[4]  = {bi, bf_, bo, bc};

    for (int t = 0; t < T_; ++t) {
        int toks[8];
#pragma unroll
        for (int i = 0; i < 8; ++i)
            toks[i] = x[(m * 16 + i + 8 * half) * T_ + t];

        v8f si[4];   // stashed sigmoid(input-gate) activations

        // Gate order: f (fold into c), i (stash), g (fold), o (produce h)
#pragma unroll
        for (int gp = 0; gp < 4; ++gp) {
            const int G = (gp == 0) ? 1 : (gp == 1) ? 0 : (gp == 2) ? 3 : 2;
            const float* __restrict__ WxG = WxArr[G];
            const float* __restrict__ bG  = bArr[G];
            const unsigned short* __restrict__ WhG = WhP + ((G * 32) << 13);

            v8f acc[4];
#pragma unroll
            for (int j = 0; j < 4; ++j)
#pragma unroll
                for (int i = 0; i < 8; ++i) acc[j][i] = 0.f;

            // h @ W_hG  — K loop over 16 chunks of 32
            for (int kc = 0; kc < 16; ++kc) {
                // A tile (16x32 bf16) from LDS:
                // lane M=ln, K = kc*32 + 8*half + {0..7, 16..23}
                BV a;
                const uint4* ap = (const uint4*)&hS[m * 16 + ln][kc * 32 + 8 * half];
                a.u[0] = ap[0];
                a.u[1] = ap[2];   // +32 bytes
#pragma unroll
                for (int j = 0; j < 4; ++j) {
                    const int nt = ng * 4 + j;
                    BV bvv;
                    const uint4* bp =
                        (const uint4*)&WhG[(((nt * 16) + kc) << 9) + lane * 16];
                    bvv.u[0] = bp[0];
                    bvv.u[1] = bp[1];
                    acc[j] = __builtin_amdgcn_wmma_f32_16x16x32_bf16(
                        false, a.v, false, bvv.v, (short)0, acc[j], false, false);
                }
            }

            // The o-pass is the last reader of hS this step; once all waves
            // finish it we may overwrite hS with the new h.
            if (gp == 3) __syncthreads();

            // Gather (one-hot matmul == fp32 row gather) + bias + activation
#pragma unroll
            for (int j = 0; j < 4; ++j) {
                const int col = ((ng * 4 + j) << 4) + ln;
                const float bias = bG[col];
#pragma unroll
                for (int i = 0; i < 8; ++i) {
                    const int row = m * 16 + i + 8 * half;
                    const float pre = acc[j][i] + WxG[toks[i] * H_ + col] + bias;
                    if (gp == 0) {                      // forget gate
                        cS[row][col] *= sigmoidf(pre);
                    } else if (gp == 1) {               // input gate
                        si[j][i] = sigmoidf(pre);
                    } else if (gp == 2) {               // cell candidate
                        cS[row][col] += si[j][i] * tanhf(pre);
                    } else {                            // output gate -> h
                        const float c = cS[row][col];
                        const float h = sigmoidf(pre) * tanhf(c);
                        const unsigned short hb = f2bf(h);
                        hS[row][col] = hb;
                        hHist[(size_t)(t * B_ + row) * H_ + col] = hb;
                        if (t == T_ - 1) {
                            outState[row * H_ + col] = h;
                            outCell[row * H_ + col]  = c;
                        }
                    }
                }
            }
        }
        __syncthreads();   // new h visible before next step's reads
    }
}

// ---------------------------------------------------------------------------
// Projection phase: out[t*B+b][v] = h_t[b] @ W_hq + b_q.
// 8192 x 10000 x 512 GEMM. One wave = 16-row m-tile x five 16-col n-tiles
// (625 n-tiles = 10000/16 exactly). 64000 waves -> 8000 blocks of 8 waves.
// ---------------------------------------------------------------------------
__global__ __launch_bounds__(256) void lstm_proj_kernel(
    const unsigned short* __restrict__ hHist,  // [8192][512] bf16 row-major
    const unsigned short* __restrict__ WhqP,   // packed [625][16][512] bf16
    const float* __restrict__ bq,              // (V)
    float* __restrict__ out)                   // (8192, V) fp32
{
    const int lane = threadIdx.x & 31;
    const int wave = threadIdx.x >> 5;
    const int wg   = blockIdx.x * 8 + wave;    // 0 .. 63999
    const int mt   = wg / 125;                 // m-tile 0..511
    const int ngrp = wg % 125;                 // n-tile group (5 tiles each)
    const int half = lane >> 4;
    const int ln   = lane & 15;

    v8f acc[5];
#pragma unroll
    for (int j = 0; j < 5; ++j)
#pragma unroll
        for (int i = 0; i < 8; ++i) acc[j][i] = 0.f;

    for (int kc = 0; kc < 16; ++kc) {
        BV a;
        const uint4* ap =
            (const uint4*)&hHist[(size_t)(mt * 16 + ln) * H_ + kc * 32 + 8 * half];
        a.u[0] = ap[0];
        a.u[1] = ap[2];   // +32 bytes
#pragma unroll
        for (int j = 0; j < 5; ++j) {
            const int nt = ngrp * 5 + j;
            BV bvv;
            const uint4* bp = (const uint4*)&WhqP[((nt * 16 + kc) << 9) + lane * 16];
            bvv.u[0] = bp[0];
            bvv.u[1] = bp[1];
            acc[j] = __builtin_amdgcn_wmma_f32_16x16x32_bf16(
                false, a.v, false, bvv.v, (short)0, acc[j], false, false);
        }
    }

#pragma unroll
    for (int j = 0; j < 5; ++j) {
        const int col = (ngrp * 5 + j) * 16 + ln;
        const float bqv = bq[col];
#pragma unroll
        for (int i = 0; i < 8; ++i) {
            const int row = mt * 16 + i + 8 * half;
            out[(size_t)row * V_ + col] = acc[j][i] + bqv;
        }
    }
}

// ---------------------------------------------------------------------------
extern "C" void kernel_launch(void* const* d_in, const int* in_sizes, int n_in,
                              void* d_out, int out_size, void* d_ws, size_t ws_size,
                              hipStream_t stream) {
    const int*   x     = (const int*)d_in[0];
    const float* state = (const float*)d_in[1];
    const float* cell  = (const float*)d_in[2];
    const float* Wxi = (const float*)d_in[3];
    const float* Whi = (const float*)d_in[4];
    const float* bi  = (const float*)d_in[5];
    const float* Wxf = (const float*)d_in[6];
    const float* Whf = (const float*)d_in[7];
    const float* bf  = (const float*)d_in[8];
    const float* Wxo = (const float*)d_in[9];
    const float* Who = (const float*)d_in[10];
    const float* bo  = (const float*)d_in[11];
    const float* Wxc = (const float*)d_in[12];
    const float* Whc = (const float*)d_in[13];
    const float* bc  = (const float*)d_in[14];
    const float* Whq = (const float*)d_in[15];
    const float* bq  = (const float*)d_in[16];

    // Workspace layout (bf16 elements)
    unsigned short* ws    = (unsigned short*)d_ws;
    unsigned short* WhP   = ws;                        // 4*32*16*512   = 1,048,576
    unsigned short* WhqP  = ws + 1048576;              // 625*16*512    = 5,120,000
    unsigned short* hHist = ws + 1048576 + 5120000;    // 128*64*512    = 4,194,304

    float* outMat   = (float*)d_out;                           // (T*B, V)
    float* outState = outMat + (size_t)T_ * B_ * V_;           // (B, H)
    float* outCell  = outState + (size_t)B_ * H_;              // (B, H)

    pack_wh_kernel<<<(1048576 + 255) / 256, 256, 0, stream>>>(Whi, Whf, Who, Whc, WhP);
    pack_whq_kernel<<<(5120000 + 255) / 256, 256, 0, stream>>>(Whq, WhqP);

    const size_t recShmem = (size_t)B_ * H_ * 2 + (size_t)B_ * H_ * 4;  // 192 KB
    lstm_recurrent_kernel<<<1, 1024, recShmem, stream>>>(
        x, state, cell, Wxi, Wxf, Wxo, Wxc, bi, bf, bo, bc,
        WhP, hHist, outState, outCell);

    lstm_proj_kernel<<<8000, 256, 0, stream>>>(hHist, WhqP, bq, outMat);
}
